// ViDAInjectGatedLinear_42975442763796
// MI455X (gfx1250) — matrix operand order repaired and checked by reading, not compiled, not run on hardware
//
#include <hip/hip_runtime.h>
#include <hip/hip_bf16.h>
#include <math.h>

typedef unsigned int u32;
typedef unsigned short u16;
typedef __attribute__((ext_vector_type(16))) __bf16 v16bf;
typedef __attribute__((ext_vector_type(2)))  __bf16 v2bf;
typedef __attribute__((ext_vector_type(2)))  float  v2f;
typedef __attribute__((ext_vector_type(8)))  float  v8f;
typedef __attribute__((address_space(3))) u16 lds16;

#define BSZ   16
#define NTOK  1024
#define CDIM  768
#define HIDD  128
#define OUTD  2304
#define MTOT  (BSZ*NTOK)        // 16384
#define STR   40                // padded LDS row stride (bf16 elems): 80B, 16B-aligned rows

// ---------- helpers ----------
union FragBF { v16bf v; u32 u[8]; };
union FragF  { v8f   v; float f[8]; };
union PkBF   { v2bf  h; u32 w; };

__device__ __forceinline__ u32 pack_bf16(float lo, float hi){
  v2f x = { lo, hi };
  PkBF p; p.h = __builtin_convertvector(x, v2bf);   // one v_cvt_pk_bf16_f32
  return p.w;
}

// A-matrix 16x32 bf16 fragment in contiguous k order (k = 16*hi + s, not the ISA table order):
// lanes 0-15: row=lane, K 0..15; lanes 16-31: row=lane-16, K 16..31 (matches frag_b)
__device__ __forceinline__ FragBF frag_a(const u16* base){
  int l = threadIdx.x & 31;
  int r = l & 15, hi = l >> 4;
  const u32* q = (const u32*)(base + r * STR + hi * 16);
  FragBF f;
#pragma unroll
  for (int i = 0; i < 8; ++i) f.u[i] = q[i];
  return f;
}
// B-matrix 32x16 bf16 fragment: lane = N col (l&15), lanes 0-15 hold K 0..15, lanes 16-31 K 16..31
__device__ __forceinline__ FragBF frag_b(const u16* base){
  int l = threadIdx.x & 31;
  int c = l & 15, hi = l >> 4;
  const u32* q = (const u32*)(base + c * STR + hi * 16);
  FragBF f;
#pragma unroll
  for (int i = 0; i < 8; ++i) f.u[i] = q[i];
  return f;
}

// ---------- WMMA GEMM: out[m,n] = sum_k A[m,k]*W[n,k] + bias[n] (+ resid[m,n]) ----------
// A: f32 [M,K] row-major (converted to bf16 while staging into LDS)
// W: bf16 [N,K] row-major (pre-converted weights; staged with async global->LDS copies)
// grid = (N/128, M/128), block = 256 (8 waves). Wave tile = 32x64 (2 A-frags x 4 B-frags
// -> 8 WMMAs, 12 DS loads/step instead of 18; each B frag feeds 2 WMMAs, each A frag 4).
// Double-buffered LDS pipeline, 2x-unrolled so buffer indices are compile-time constants.
// Requires K % 64 == 0 (all K here: 768, 128).
__global__ __launch_bounds__(256) void gemm_bf16_wmma(
    const float* __restrict__ A, const u16* __restrict__ Wbf,
    const float* __restrict__ bias, const float* __restrict__ resid,
    float* __restrict__ out, int M, int N, int K)
{
  __shared__ u16 ldsA[2][128 * STR];
  __shared__ u16 ldsW[2][128 * STR];
  const int m0 = blockIdx.y * 128;
  const int n0 = blockIdx.x * 128;
  const int t    = threadIdx.x;
  const int row  = t >> 1;        // 0..127: tile row this thread stages (A path)
  const int hf   = t & 1;         // which 16-wide half of the 32-wide K slab
  const int wv   = t >> 5;        // wave id 0..7
  const int lane = t & 31;
  const int wm   = (wv >> 1) * 32;  // wave M offset in tile: 0,32,64,96
  const int wn   = (wv & 1) * 64;   // wave N offset in tile: 0,64

  FragF acc[2][4];
#pragma unroll
  for (int i = 0; i < 2; ++i)
#pragma unroll
    for (int j = 0; j < 4; ++j)
#pragma unroll
      for (int r = 0; r < 8; ++r) acc[i][j].f[r] = 0.0f;

  float fa[16];                   // in-flight A slab (f32)

  // issue global f32 loads for an A slab
  auto loadA = [&](int kt){
    const float* ga = A + (size_t)(m0 + row) * K + kt + hf * 16;
#pragma unroll
    for (int i = 0; i < 4; ++i) ((float4*)fa)[i] = ((const float4*)ga)[i];
  };
  // convert + store the in-flight A slab into LDS buffer `buf`
  auto storeA = [&](int buf){
    u32* da = (u32*)(&ldsA[buf][row * STR + hf * 16]);
#pragma unroll
    for (int i = 0; i < 8; ++i) da[i] = pack_bf16(fa[2 * i], fa[2 * i + 1]);
  };
  // async global->LDS copy of a W slab (raw bf16 bytes); 2 x 16B chunks per thread
  auto asyncW = [&](int kt, int buf){
#pragma unroll
    for (int i = 0; i < 2; ++i){
      int ch  = t * 2 + i;        // 0..511 chunks of 16B; 4 chunks per 32-elem row
      int r   = ch >> 2, sub = ch & 3;
      u32 ldsoff = (u32)(unsigned long long)(lds16*)&ldsW[buf][r * STR + sub * 8];
      u32 goff   = (u32)(((size_t)(n0 + r) * K + kt + sub * 8) * 2);
      asm volatile("global_load_async_to_lds_b128 %0, %1, %2"
                   :: "v"(ldsoff), "v"(goff), "s"(Wbf) : "memory");
    }
  };
  // 32x64 wave tile: 2 A frags, 4 B frags, 8 WMMAs; every DS-loaded frag is reused.
  auto compute = [&](int buf){
    FragBF a0 = frag_a(&ldsA[buf][(wm) * STR]);
    FragBF a1 = frag_a(&ldsA[buf][(wm + 16) * STR]);
#pragma unroll
    for (int j = 0; j < 4; ++j){
      FragBF b = frag_b(&ldsW[buf][(wn + j * 16) * STR]);
      acc[0][j].v = __builtin_amdgcn_wmma_f32_16x16x32_bf16(
          false, a0.v, false, b.v, (short)0, acc[0][j].v, false, false);
      acc[1][j].v = __builtin_amdgcn_wmma_f32_16x16x32_bf16(
          false, a1.v, false, b.v, (short)0, acc[1][j].v, false, false);
    }
  };

  const int steps = K >> 5;       // even for all shapes here

  // prologue: fill buffer 0
  loadA(0);
  asyncW(0, 0);
  storeA(0);
  asm volatile("s_wait_asynccnt 0" ::: "memory");
  __syncthreads();

  for (int ks = 0; ks < steps; ks += 2){
    // ---- even step: compute buf0, stage buf1 ----
    loadA((ks + 1) << 5);
    asyncW((ks + 1) << 5, 1);
    compute(0);
    storeA(1);
    asm volatile("s_wait_asynccnt 0" ::: "memory");
    __syncthreads();

    // ---- odd step: compute buf1, stage buf0 ----
    const bool more = (ks + 2) < steps;
    if (more){
      loadA((ks + 2) << 5);
      asyncW((ks + 2) << 5, 0);
    }
    compute(1);
    if (more){
      storeA(0);
      asm volatile("s_wait_asynccnt 0" ::: "memory");
    }
    __syncthreads();
  }

  // C/D layout: VGPR r, lanes 0-15 -> M=r, lanes 16-31 -> M=8+r; N = lane&15
  const int rsel = lane >> 4, ncol = lane & 15;
#pragma unroll
  for (int i = 0; i < 2; ++i)
#pragma unroll
  for (int j = 0; j < 4; ++j){
    const int n  = n0 + wn + j * 16 + ncol;
    const float bb = bias ? bias[n] : 0.0f;
#pragma unroll
    for (int r = 0; r < 8; ++r){
      const int m = m0 + wm + i * 16 + r + rsel * 8;
      float v = acc[i][j].f[r] + bb;
      if (resid) v += resid[(size_t)m * N + n];
      out[(size_t)m * N + n] = v;
    }
  }
}

// ---------- f32 -> bf16 weight conversion ----------
__global__ void cvt_bf16(const float* __restrict__ in, u16* __restrict__ out, int n){
  int i = blockIdx.x * 256 + threadIdx.x;
  if (i < n){
    u32 w = pack_bf16(in[i], 0.0f);
    out[i] = (u16)(w & 0xFFFFu);
  }
}

// ---------- depthwise 3x3 conv (SAME, zero pad) + bias + exact GELU ----------
__global__ __launch_bounds__(256) void dwconv_gelu(
    const float* __restrict__ X1, const float* __restrict__ dw_w,
    const float* __restrict__ dw_b, float* __restrict__ act)
{
  int idx = blockIdx.x * 256 + threadIdx.x;   // [0, 16*1024*128)
  int c = idx & 127;
  int n = (idx >> 7) & 1023;
  int b = idx >> 17;
  int h = n >> 5, w = n & 31;
  float s = dw_b[c];
#pragma unroll
  for (int kh = 0; kh < 3; ++kh){
    int hh = h + kh - 1;
    if (hh < 0 || hh > 31) continue;
#pragma unroll
    for (int kw = 0; kw < 3; ++kw){
      int ww = w + kw - 1;
      if (ww < 0 || ww > 31) continue;
      s += X1[((size_t)b * 1024 + (hh << 5) + ww) * 128 + c] * dw_w[c * 9 + kh * 3 + kw];
    }
  }
  act[idx] = 0.5f * s * (1.0f + erff(s * 0.70710678118654752f));
}

// ---------- per-batch token mean of `input` ----------
__global__ void mean_kernel(const float* __restrict__ input, float* __restrict__ mean){
  int blk = blockIdx.x;            // 48 = 16 batches * 3 chunks of 256 channels
  int b = blk / 3, part = blk % 3;
  int c = part * 256 + threadIdx.x;
  const float* p = input + (size_t)b * NTOK * CDIM + c;
  float s = 0.f;
  for (int n = 0; n < NTOK; ++n) s += p[(size_t)n * CDIM];
  mean[b * CDIM + c] = s * (1.0f / 1024.0f);
}

__global__ void norm_kernel(const float* __restrict__ mean, float* __restrict__ nrm){
  __shared__ float red[256];
  int b = blockIdx.x;
  float s = 0.f;
  for (int c = threadIdx.x; c < CDIM; c += 256){ float v = mean[b * CDIM + c]; s += v * v; }
  red[threadIdx.x] = s; __syncthreads();
  for (int o = 128; o > 0; o >>= 1){
    if (threadIdx.x < (unsigned)o) red[threadIdx.x] += red[threadIdx.x + o];
    __syncthreads();
  }
  if (threadIdx.x == 0) nrm[b] = sqrtf(red[0]);
}

// ---------- cosine similarity, one wave32 per token ----------
__global__ __launch_bounds__(256) void csim_kernel(
    const float* __restrict__ refine, const float* __restrict__ mean,
    const float* __restrict__ nrm, float* __restrict__ csim)
{
  int tok  = blockIdx.x * 8 + (threadIdx.x >> 5);
  int lane = threadIdx.x & 31;
  int b = tok >> 10;
  const float* x = refine + (size_t)tok * CDIM;
  const float* m = mean + b * CDIM;
  float dot = 0.f, nn = 0.f;
  for (int c = lane; c < CDIM; c += 32){ float v = x[c]; dot += v * m[c]; nn += v * v; }
#pragma unroll
  for (int o = 16; o > 0; o >>= 1){ dot += __shfl_xor(dot, o, 32); nn += __shfl_xor(nn, o, 32); }
  if (lane == 0) csim[tok] = dot / (nrm[b] * sqrtf(nn));
}

__global__ void minmax_kernel(const float* __restrict__ csim, float* __restrict__ mnmx){
  __shared__ float smn[256], smx[256];
  int b = blockIdx.x, tid = threadIdx.x;
  float mn = 3.4e38f, mx = -3.4e38f;
  for (int n = tid; n < NTOK; n += 256){
    float v = csim[b * NTOK + n]; mn = fminf(mn, v); mx = fmaxf(mx, v);
  }
  smn[tid] = mn; smx[tid] = mx; __syncthreads();
  for (int o = 128; o > 0; o >>= 1){
    if (tid < o){ smn[tid] = fminf(smn[tid], smn[tid + o]); smx[tid] = fmaxf(smx[tid], smx[tid + o]); }
    __syncthreads();
  }
  if (tid == 0){ mnmx[2 * b] = smn[0]; mnmx[2 * b + 1] = smx[0]; }
}

// ---------- ThreeFry-2x32-20 (JAX), key = (0, 42) ----------
__device__ __forceinline__ u32 rotl32(u32 x, int d){ return (x << d) | (x >> (32 - d)); }
__device__ __forceinline__ void threefry2x32(u32 c0, u32 c1, u32& o0, u32& o1){
  const u32 ks0 = 0u, ks1 = 42u, ks2 = 0u ^ 42u ^ 0x1BD11BDAu;
  const u32 ks[3] = { ks0, ks1, ks2 };
  const int rot[2][4] = { {13, 15, 26, 6}, {17, 29, 16, 24} };
  u32 x0 = c0 + ks0, x1 = c1 + ks1;
#pragma unroll
  for (int i = 0; i < 5; ++i){
#pragma unroll
    for (int r = 0; r < 4; ++r){ x0 += x1; x1 = rotl32(x1, rot[i & 1][r]); x1 ^= x0; }
    x0 += ks[(i + 1) % 3];
    x1 += ks[(i + 2) % 3] + (u32)(i + 1);
  }
  o0 = x0; o1 = x1;
}
__device__ __forceinline__ u32 gumbel_bits(u32 e){
  // jax _random_bits: counts = arange(32768); x0 = counts[:16384], x1 = counts[16384:]
  u32 o0, o1;
  if (e < 16384u){ threefry2x32(e, e + 16384u, o0, o1); return o0; }
  threefry2x32(e - 16384u, e, o0, o1); return o1;
}
__device__ __forceinline__ float bits_to_gumbel(u32 bits){
  const float TINY = 1.17549435e-38f;
  float f = __uint_as_float((bits >> 9) | 0x3f800000u) - 1.0f;   // [0,1)
  float u = f * (1.0f - TINY) + TINY;
  u = fmaxf(TINY, u);
  return -logf(-logf(u));
}

// ---------- router: minmax-normalize csim -> 1->128->2 MLP -> sigmoid -> gumbel softmax[:,0] ----------
__global__ __launch_bounds__(256) void router_kernel(
    const float* __restrict__ csim, const float* __restrict__ mnmx,
    const float* __restrict__ r1_w, const float* __restrict__ r1_b,
    const float* __restrict__ r2_w, const float* __restrict__ r2_b,
    float* __restrict__ out_router)
{
  int t = blockIdx.x * 256 + threadIdx.x;   // token 0..16383
  int b = t >> 10;
  float mn = mnmx[2 * b], mx = mnmx[2 * b + 1];
  float x = csim[t];
  float rng = mx - mn;
  if (rng != 0.0f) x = (x - mn) / rng;
  float l0 = r2_b[0], l1 = r2_b[1];
  for (int j = 0; j < HIDD; ++j){
    float h = fmaxf(0.0f, r1_w[j] * x + r1_b[j]);
    l0 += r2_w[j] * h;
    l1 += r2_w[HIDD + j] * h;
  }
  l0 = 1.0f / (1.0f + expf(-l0));
  l1 = 1.0f / (1.0f + expf(-l1));
  float g0 = bits_to_gumbel(gumbel_bits((u32)(2 * t)));
  float g1 = bits_to_gumbel(gumbel_bits((u32)(2 * t + 1)));
  // softmax over 2, take index 0
  out_router[t] = 1.0f / (1.0f + expf((l1 + g1) - (l0 + g0)));
}

// ---------- launch ----------
static inline size_t align256(size_t x){ return (x + 255) & ~(size_t)255; }

extern "C" void kernel_launch(void* const* d_in, const int* in_sizes, int n_in,
                              void* d_out, int out_size, void* d_ws, size_t ws_size,
                              hipStream_t stream) {
  const float* input  = (const float*)d_in[0];
  const float* input2 = (const float*)d_in[3];
  const float* qkv_w  = (const float*)d_in[4];
  const float* qkv_b  = (const float*)d_in[5];
  const float* fc1_w  = (const float*)d_in[6];
  const float* fc1_b  = (const float*)d_in[7];
  const float* dw_w   = (const float*)d_in[8];
  const float* dw_b   = (const float*)d_in[9];
  const float* fc2_w  = (const float*)d_in[10];
  const float* fc2_b  = (const float*)d_in[11];
  const float* r1_w   = (const float*)d_in[12];
  const float* r1_b   = (const float*)d_in[13];
  const float* r2_w   = (const float*)d_in[14];
  const float* r2_b   = (const float*)d_in[15];

  float* q1_out  = (float*)d_out;
  float* q2_out  = q1_out + (size_t)MTOT * OUTD;
  float* rt_out  = q2_out + (size_t)MTOT * OUTD;

  // workspace carve-up
  char* w = (char*)d_ws;
  size_t off = 0;
  u16* Wqkv_bf = (u16*)(w + off); off = align256(off + (size_t)OUTD * CDIM * 2);
  u16* Wfc1_bf = (u16*)(w + off); off = align256(off + (size_t)HIDD * CDIM * 2);
  u16* Wfc2_bf = (u16*)(w + off); off = align256(off + (size_t)CDIM * HIDD * 2);
  float* X1    = (float*)(w + off); off = align256(off + (size_t)MTOT * HIDD * 4);
  float* act   = (float*)(w + off); off = align256(off + (size_t)MTOT * HIDD * 4);
  float* refine= (float*)(w + off); off = align256(off + (size_t)MTOT * CDIM * 4);
  float* meanb = (float*)(w + off); off = align256(off + (size_t)BSZ * CDIM * 4);
  float* nrmb  = (float*)(w + off); off = align256(off + (size_t)BSZ * 4);
  float* csim  = (float*)(w + off); off = align256(off + (size_t)MTOT * 4);
  float* mnmx  = (float*)(w + off); off = align256(off + (size_t)BSZ * 2 * 4);
  (void)ws_size; (void)in_sizes; (void)n_in; (void)out_size;

  // 1) convert weights to bf16 once
  cvt_bf16<<<(OUTD * CDIM + 255) / 256, 256, 0, stream>>>(qkv_w, Wqkv_bf, OUTD * CDIM);
  cvt_bf16<<<(HIDD * CDIM + 255) / 256, 256, 0, stream>>>(fc1_w, Wfc1_bf, HIDD * CDIM);
  cvt_bf16<<<(CDIM * HIDD + 255) / 256, 256, 0, stream>>>(fc2_w, Wfc2_bf, CDIM * HIDD);

  // 2) q1 = input @ qkv_w^T + qkv_b
  gemm_bf16_wmma<<<dim3(OUTD / 128, MTOT / 128), 256, 0, stream>>>(
      input, Wqkv_bf, qkv_b, nullptr, q1_out, MTOT, OUTD, CDIM);

  // 3) fc1: X1 = input2 @ fc1_w^T + fc1_b   [16384,128]
  gemm_bf16_wmma<<<dim3(HIDD / 128, MTOT / 128), 256, 0, stream>>>(
      input2, Wfc1_bf, fc1_b, nullptr, X1, MTOT, HIDD, CDIM);

  // 4) depthwise 3x3 + bias + GELU(erf)
  dwconv_gelu<<<(MTOT * HIDD) / 256, 256, 0, stream>>>(X1, dw_w, dw_b, act);

  // 5) fc2 + residual: refine = input2 + act @ fc2_w^T + fc2_b
  gemm_bf16_wmma<<<dim3(CDIM / 128, MTOT / 128), 256, 0, stream>>>(
      act, Wfc2_bf, fc2_b, input2, refine, MTOT, CDIM, HIDD);

  // 6) q2 = refine @ qkv_w^T + qkv_b
  gemm_bf16_wmma<<<dim3(OUTD / 128, MTOT / 128), 256, 0, stream>>>(
      refine, Wqkv_bf, qkv_b, nullptr, q2_out, MTOT, OUTD, CDIM);

  // 7) router path
  mean_kernel<<<BSZ * 3, 256, 0, stream>>>(input, meanb);
  norm_kernel<<<BSZ, 256, 0, stream>>>(meanb, nrmb);
  csim_kernel<<<MTOT / 8, 256, 0, stream>>>(refine, meanb, nrmb, csim);
  minmax_kernel<<<BSZ, 256, 0, stream>>>(csim, mnmx);
  router_kernel<<<MTOT / 256, 256, 0, stream>>>(csim, mnmx, r1_w, r1_b, r2_w, r2_b, rt_out);
}
